// DNM_81312320848611
// MI455X (gfx1250) — matrix operand order repaired and checked by reading, not compiled, not run on hardware
//
#include <hip/hip_runtime.h>
#include <hip/hip_bf16.h>
#include <stdint.h>

// ---------------------------------------------------------------------------
// KPN U-Net on gfx1250: implicit-GEMM conv via v_wmma_f32_16x16x32_bf16.
// Compute-bound (~0.42 TFLOP GEMM work vs ~1-2 GB traffic) -> WMMA path.
// Block tile: 64 couts x 64 pixels, K-step 64 -> 8 WMMAs per wave per stage.
// Weight tiles are DMA'd into LDS by the Tensor Data Mover (TENSORcnt),
// overlapping with the on-the-fly im2col gather done by the VALU.
// All spatial dims are powers of two -> shift/mask addressing, no divides.
// ---------------------------------------------------------------------------

typedef __attribute__((ext_vector_type(16))) __bf16 bf16x16;
typedef __attribute__((ext_vector_type(8)))  float  v8f;
typedef __attribute__((ext_vector_type(4)))  unsigned int u32x4;
typedef __attribute__((ext_vector_type(8)))  int   i32x8;
typedef __attribute__((ext_vector_type(4)))  int   i32x4;

#if defined(__has_builtin)
#if __has_builtin(__builtin_amdgcn_tensor_load_to_lds)
#define HAVE_TDM 1
#endif
#endif

__device__ inline unsigned short f32_to_bf16_bits(float f) {
    union { float f; unsigned int u; } x; x.f = f;
    unsigned int u = x.u;
    unsigned int r = u + 0x7fffu + ((u >> 16) & 1u);   // round-to-nearest-even
    return (unsigned short)(r >> 16);
}

__device__ inline v8f wmma_bf16(const bf16x16& a, const bf16x16& b, v8f c) {
    return __builtin_amdgcn_wmma_f32_16x16x32_bf16(
        false, a, false, b, (short)0, c, false, false);
}

#ifdef HAVE_TDM
// ---------------------------------------------------------------------------
// TDM: DMA a 2D tile (tile1 rows x tile0 elements, 2-byte elements, row
// stride = stride0 elements) from global into LDS at lds_addr.
// D# layout per CDNA5 ISA 8.3/8.4 (group0: count/lds/global/type,
// group1: data_size, tensor dims, tile dims, dim0 stride).
// ---------------------------------------------------------------------------
__device__ inline void tdm_load_2d_b16(unsigned lds_addr, const void* gptr,
                                       unsigned dim0, unsigned dim1,
                                       unsigned tile0, unsigned tile1,
                                       unsigned stride0) {
    unsigned long long ga = (unsigned long long)(uintptr_t)gptr;
    u32x4 g0;
    g0[0] = 1u;                                           // count=1, user D#
    g0[1] = lds_addr;                                     // LDS byte address
    g0[2] = (unsigned)(ga & 0xffffffffu);                 // global addr lo
    g0[3] = (unsigned)((ga >> 32) & 0x01ffffffu) | (2u << 30); // addr hi | type=2
    i32x8 g1;
    g1[0] = (int)(1u << 16);                              // data_size=1 (2 bytes)
    g1[1] = (int)((dim0 & 0xffffu) << 16);                // tensor_dim0 lo16
    g1[2] = (int)((dim0 >> 16) | ((dim1 & 0xffffu) << 16)); // dim0 hi / dim1 lo
    g1[3] = (int)((dim1 >> 16) | (tile0 << 16));          // dim1 hi / tile_dim0
    g1[4] = (int)(tile1 & 0xffffu);                       // tile_dim1, tile_dim2=0
    g1[5] = (int)stride0;                                 // tensor_dim0_stride lo32
    g1[6] = 0;                                            // stride hi16, dim1_stride
    g1[7] = 0;
    i32x4 gz = {0, 0, 0, 0};                              // 2D: groups 2/3 unused
#if __clang_major__ >= 23
    i32x8 gz8 = {0, 0, 0, 0, 0, 0, 0, 0};
    __builtin_amdgcn_tensor_load_to_lds(g0, g1, gz, gz, gz8, 0);
#else
    __builtin_amdgcn_tensor_load_to_lds(g0, g1, gz, gz, 0);
#endif
}
#endif

// ---------------------------------------------------------------------------
// Weight prep: OIHW fp32 -> bf16 GEMM layout [CoutPad][Kpad] (k contiguous),
// zero padded. k = ci*taps + tap. K-contiguous rows make the TDM tile land
// in LDS in the [co][k] layout the A fragments read.
// ---------------------------------------------------------------------------
__global__ void wprep_kernel(const float* __restrict__ w, unsigned short* __restrict__ out,
                             int cin, int cout, int ks, int kpad, int coutpad) {
    int idx = blockIdx.x * blockDim.x + threadIdx.x;
    int total = kpad * coutpad;
    if (idx >= total) return;
    int co = idx / kpad;
    int k  = idx - co * kpad;
    int taps = ks * ks;
    float v = 0.0f;
    if (co < cout && k < cin * taps) {
        int ci = k / taps, t = k - ci * taps;
        v = w[((size_t)co * cin + ci) * taps + t];
    }
    out[idx] = f32_to_bf16_bits(v);
}

// ---------------------------------------------------------------------------
// Implicit-GEMM conv, KS in {1,3} at compile time. Optional ReLU, optional
// fused two-source concat (channels [0,C0) from x0, [C0,Cin) from x1).
//
// Block: 128 threads = 4 waves. Block tile: 64 couts x 64 pixels.
// Per K-step (64): TDM stages A = 64x64 bf16 weights into LDS (wave 0 issues,
// s_wait_tensorcnt fences); all waves gather B = 64x64 bf16 im2col. Each wave
// then runs 8 x v_wmma_f32_16x16x32_bf16 (4 cout tiles x 2 K-chunks, B
// fragments shared across cout tiles).
// ---------------------------------------------------------------------------
template <int KS>
__global__ __launch_bounds__(128)
void conv_wmma_kernel(const float* __restrict__ x0, const float* __restrict__ x1,
                      int C0, int Cin,
                      const unsigned short* __restrict__ wbf,
                      const float* __restrict__ bias,
                      float* __restrict__ y,
                      int lw, int lhw, int Cout, int CoutPad,
                      int Kpad, int relu) {
    constexpr int taps = KS * KS;
    constexpr int pad  = KS / 2;
    __shared__ unsigned short lA[64 * 64];   // [co][k]  : k contiguous
    __shared__ unsigned short lB[64 * 64];   // [pix][k] : k contiguous

    const int tid  = threadIdx.x;
    const int wave = tid >> 5;
    const int lane = tid & 31;
    const int cotile  = blockIdx.x;
    const int pixtile = blockIdx.y;

    const int Wd = 1 << lw;
    const int H  = 1 << (lhw - lw);
    const int HW = 1 << lhw;
    const int C1 = Cin - C0;
    const int p0 = pixtile * 64;
    const int b  = p0 >> lhw;           // whole tile in one image
    const int hw0 = p0 & (HW - 1);
    const float* xb0 = x0 + ((size_t)b * C0 << lhw);
    const float* xb1 = x1 + ((size_t)b * C1 << lhw);
    const int cobase = cotile * 64;

    v8f acc[4];
    #pragma unroll
    for (int c = 0; c < 4; ++c) acc[c] = (v8f){};

    for (int ko = 0; ko < Kpad; ko += 64) {
        // ---- stage A (weights 64co x 64k) via Tensor Data Mover ----
#ifdef HAVE_TDM
        if (wave == 0) {
            tdm_load_2d_b16((unsigned)(uintptr_t)&lA[0],
                            wbf + (size_t)cobase * Kpad + ko,
                            (unsigned)Kpad, (unsigned)CoutPad,
                            64u, 64u, (unsigned)Kpad);
        }
#else
        #pragma unroll
        for (int e = 0; e < 32; ++e) {
            int idx = tid + e * 128;
            int kl = idx & 63, co = idx >> 6;
            lA[co * 64 + kl] = wbf[(size_t)(cobase + co) * Kpad + ko + kl];
        }
#endif
        // ---- stage B (on-the-fly im2col): 64x64 = 4096 elems, 32/thread
        #pragma unroll
        for (int e = 0; e < 32; ++e) {
            int idx = tid + e * 128;
            int pix = idx >> 6, kl = idx & 63;
            int k = ko + kl;
            float v = 0.0f;
            int ci = k / taps;                      // compile-time taps -> magic mul
            if (ci < Cin) {
                int t  = k - ci * taps;
                int hw = hw0 + pix;
                int h  = hw >> lw;
                int w  = hw & (Wd - 1);
                int h2 = h + t / KS - pad;
                int w2 = w + t % KS - pad;
                if (((unsigned)h2 < (unsigned)H) & ((unsigned)w2 < (unsigned)Wd)) {
                    v = (ci < C0)
                        ? xb0[((size_t)ci << lhw) + (h2 << lw) + w2]
                        : xb1[((size_t)(ci - C0) << lhw) + (h2 << lw) + w2];
                }
            }
            lB[pix * 64 + kl] = f32_to_bf16_bits(v);
        }
#ifdef HAVE_TDM
        if (wave == 0) __builtin_amdgcn_s_wait_tensorcnt(0);
#endif
        __syncthreads();

        // ---- fragments: A and B share contiguous k = 16*hi + i order (wave32)
        const int m  = lane & 15;
        const int hi = lane >> 4;
        const int pp = wave * 16 + m;
        union { bf16x16 v; unsigned short u[16]; } af[4], bf[2];
        // B 32x16: lanes 0-15 -> N=lane, K c..c+15; lanes 16-31 -> K c+16..c+31
        #pragma unroll
        for (int q = 0; q < 2; ++q)
            #pragma unroll
            for (int i = 0; i < 16; ++i)
                bf[q].u[i] = lB[pp * 64 + q * 32 + hi * 16 + i];
        // A 16x32: lanes 0-15 hold K c..c+15,
        //          lanes 16-31 hold K c+16..c+31
        #pragma unroll
        for (int q = 0; q < 2; ++q) {
            #pragma unroll
            for (int c = 0; c < 4; ++c) {
                const int row = (c * 16 + m) * 64 + q * 32;
                #pragma unroll
                for (int i = 0; i < 8; ++i) {
                    af[c].u[i]     = lA[row + hi * 16 + i];
                    af[c].u[8 + i] = lA[row + hi * 16 + 8 + i];
                }
            }
            acc[0] = wmma_bf16(af[0].v, bf[q].v, acc[0]);
            acc[1] = wmma_bf16(af[1].v, bf[q].v, acc[1]);
            acc[2] = wmma_bf16(af[2].v, bf[q].v, acc[2]);
            acc[3] = wmma_bf16(af[3].v, bf[q].v, acc[3]);
        }
        __syncthreads();
    }

    // ---- epilogue: D layout -> VGPR r: M = r (+8 for lanes>=16), N = lane&15
    int off = hw0 + wave * 16 + (lane & 15);
    #pragma unroll
    for (int c = 0; c < 4; ++c) {
        #pragma unroll
        for (int r = 0; r < 8; ++r) {
            int col = (lane < 16) ? r : (r + 8);
            int co  = cobase + c * 16 + col;
            if (co < Cout) {
                float val = acc[c][r] + bias[co];
                if (relu) val = fmaxf(val, 0.0f);
                y[(((size_t)b * Cout + co) << lhw) + off] = val;
            }
        }
    }
}

// ---------------------------------------------------------------------------
// 2x2 average pool
// ---------------------------------------------------------------------------
__global__ void avgpool2_kernel(const float* __restrict__ x, float* __restrict__ y,
                                int BC, int H, int W) {
    int Ho = H >> 1, Wo = W >> 1;
    size_t total = (size_t)BC * Ho * Wo;
    size_t i = (size_t)blockIdx.x * blockDim.x + threadIdx.x;
    if (i >= total) return;
    int wo = (int)(i % Wo);
    size_t r = i / Wo;
    int ho = (int)(r % Ho);
    size_t bc = r / Ho;
    const float* p = x + (bc * H + (size_t)ho * 2) * W + (size_t)wo * 2;
    y[i] = 0.25f * (p[0] + p[1] + p[W] + p[W + 1]);
}

// ---------------------------------------------------------------------------
// Bilinear 2x upsample, half-pixel (align_corners=False)
// ---------------------------------------------------------------------------
__global__ void up2_kernel(const float* __restrict__ x, float* __restrict__ y,
                           int BC, int H, int W) {
    int Ho = H * 2, Wo = W * 2;
    size_t total = (size_t)BC * Ho * Wo;
    size_t i = (size_t)blockIdx.x * blockDim.x + threadIdx.x;
    if (i >= total) return;
    int wo = (int)(i % Wo);
    size_t r = i / Wo;
    int ho = (int)(r % Ho);
    size_t bc = r / Ho;
    float fy = (ho + 0.5f) * 0.5f - 0.5f;
    float fx = (wo + 0.5f) * 0.5f - 0.5f;
    int y0i = (int)floorf(fy), x0i = (int)floorf(fx);
    float ay = fy - y0i, ax = fx - x0i;
    int y1i = min(H - 1, max(0, y0i + 1));
    int x1i = min(W - 1, max(0, x0i + 1));
    y0i = min(H - 1, max(0, y0i));
    x0i = min(W - 1, max(0, x0i));
    const float* p = x + bc * H * W;
    float v = (1.0f - ay) * ((1.0f - ax) * p[(size_t)y0i * W + x0i] + ax * p[(size_t)y0i * W + x1i])
            +         ay  * ((1.0f - ax) * p[(size_t)y1i * W + x0i] + ax * p[(size_t)y1i * W + x1i]);
    y[i] = v;
}

// ---------------------------------------------------------------------------
// Per-pixel 5x5 dynamic convolution: out[b,c,h,w] = sum_t core[b,t*3+c,h,w] *
// data_padded[b,c,h+dy,w+dx]   (t = i*5+j, dy=i-2, dx=j-2)
// ---------------------------------------------------------------------------
__global__ void kpn_apply_kernel(const float* __restrict__ data,
                                 const float* __restrict__ core,
                                 float* __restrict__ out,
                                 int B, int C, int H, int W) {
    size_t total = (size_t)B * C * H * W;
    size_t i = (size_t)blockIdx.x * blockDim.x + threadIdx.x;
    if (i >= total) return;
    int w = (int)(i % W);
    size_t r = i / W;
    int h = (int)(r % H);
    r /= H;
    int c = (int)(r % C);
    int b = (int)(r / C);
    size_t HW = (size_t)H * W;
    float s = 0.0f;
    #pragma unroll
    for (int t = 0; t < 25; ++t) {
        int h2 = h + t / 5 - 2;
        int w2 = w + t % 5 - 2;
        float dv = 0.0f;
        if (h2 >= 0 && h2 < H && w2 >= 0 && w2 < W)
            dv = data[((size_t)b * C + c) * HW + (size_t)h2 * W + w2];
        float cv = core[((size_t)b * 75 + t * 3 + c) * HW + (size_t)h * W + w];
        s += dv * cv;
    }
    out[i] = s;
}

// ---------------------------------------------------------------------------
// Host orchestration
// ---------------------------------------------------------------------------
extern "C" void kernel_launch(void* const* d_in, const int* in_sizes, int n_in,
                              void* d_out, int out_size, void* d_ws, size_t ws_size,
                              hipStream_t stream) {
    (void)in_sizes; (void)n_in; (void)out_size; (void)ws_size;

    const int B = 4;
    const float* data = (const float*)d_in[0];

    // 24 3x3 layers (conv1..conv8, 3 each) + outc 1x1
    static const int cins[25]  = {   3,  64,  64,   64, 128, 128,  128, 256, 256,
                                   256, 512, 512,  512, 512, 512, 1024, 512, 512,
                                   768, 256, 256,  384,  75,  75,   75};
    static const int couts[25] = {  64,  64,  64,  128, 128, 128,  256, 256, 256,
                                   512, 512, 512,  512, 512, 512,  512, 512, 512,
                                   256, 256, 256,   75,  75,  75,   75};
    const float* W[25];
    const float* Bi[25];
    for (int i = 0; i < 24; ++i) { W[i] = (const float*)d_in[1 + 2 * i]; Bi[i] = (const float*)d_in[2 + 2 * i]; }
    W[24] = (const float*)d_in[49]; Bi[24] = (const float*)d_in[50];

    int ksz[25], kpads[25], copads[25];
    for (int i = 0; i < 25; ++i) {
        ksz[i]   = (i == 24) ? 1 : 3;
        int K    = cins[i] * ksz[i] * ksz[i];
        kpads[i] = (K + 63) & ~63;          // K-step 64
        copads[i]= (couts[i] + 63) & ~63;   // 64-cout block tile
    }

    // bump allocator in d_ws
    char* ws = (char*)d_ws;
    size_t cur = 0;
    auto alloc = [&](size_t bytes) -> void* {
        void* p = ws + cur;
        cur = (cur + bytes + 255) & ~(size_t)255;
        return p;
    };

    // weights -> bf16 GEMM layout [CoutPad][Kpad] (k contiguous, TDM-friendly)
    unsigned short* wbf[25];
    for (int i = 0; i < 25; ++i) {
        size_t n = (size_t)kpads[i] * copads[i];
        wbf[i] = (unsigned short*)alloc(n * 2);
        int blocks = (int)((n + 255) / 256);
        wprep_kernel<<<blocks, 256, 0, stream>>>(W[i], wbf[i], cins[i], couts[i],
                                                 ksz[i], kpads[i], copads[i]);
    }

    auto conv = [&](const float* x0, const float* x1, int C0, int li,
                    int H, int Wd, float* y, int relu) {
        int lw  = __builtin_ctz(Wd);
        int lhw = __builtin_ctz(H * Wd);
        dim3 grid(copads[li] / 64, (unsigned)((B * H * Wd) / 64));
        const float* xx1 = x1 ? x1 : x0;
        int c0 = x1 ? C0 : cins[li];
        if (ksz[li] == 3) {
            conv_wmma_kernel<3><<<grid, 128, 0, stream>>>(
                x0, xx1, c0, cins[li], wbf[li], Bi[li], y,
                lw, lhw, couts[li], copads[li], kpads[li], relu);
        } else {
            conv_wmma_kernel<1><<<grid, 128, 0, stream>>>(
                x0, xx1, c0, cins[li], wbf[li], Bi[li], y,
                lw, lhw, couts[li], copads[li], kpads[li], relu);
        }
    };
    auto pool = [&](const float* x, float* y, int C, int H, int Wd) {
        size_t n = (size_t)B * C * (H / 2) * (Wd / 2);
        avgpool2_kernel<<<(unsigned)((n + 255) / 256), 256, 0, stream>>>(x, y, B * C, H, Wd);
    };
    auto up = [&](const float* x, float* y, int C, int H, int Wd) {
        size_t n = (size_t)B * C * (H * 2) * (Wd * 2);
        up2_kernel<<<(unsigned)((n + 255) / 256), 256, 0, stream>>>(x, y, B * C, H, Wd);
    };

    auto fbuf = [&](int C, int HW) -> float* { return (float*)alloc((size_t)B * C * HW * 4); };

    // ---- activations ----
    float* a1 = fbuf(64, 65536);          // 256^2 ping
    float* a2 = fbuf(64, 65536);          // 256^2 pong
    float* p1 = fbuf(64, 16384);
    float* t128a = fbuf(128, 16384);
    float* t128b = fbuf(128, 16384);
    float* c2 = fbuf(128, 16384);         // skip (live until conv8)
    float* p2 = fbuf(128, 4096);
    float* t64a = fbuf(256, 4096);
    float* t64b = fbuf(256, 4096);
    float* c3 = fbuf(256, 4096);          // skip
    float* p3 = fbuf(256, 1024);
    float* t32a = fbuf(512, 1024);
    float* t32b = fbuf(512, 1024);
    float* c4 = fbuf(512, 1024);          // skip
    float* p4 = fbuf(512, 256);
    float* t16a = fbuf(512, 256);
    float* t16b = fbuf(512, 256);
    float* u6 = fbuf(512, 16384);
    float* u7 = fbuf(256, 65536);
    float* u8 = fbuf(75, 262144);
    float* core = fbuf(75, 262144);

    // conv1 @256
    conv(data, nullptr, 0, 0, 256, 256, a1, 1);
    conv(a1, nullptr, 0, 1, 256, 256, a2, 1);
    conv(a2, nullptr, 0, 2, 256, 256, a1, 1);       // c1 = a1
    pool(a1, p1, 64, 256, 256);
    // conv2 @128
    conv(p1, nullptr, 0, 3, 128, 128, t128a, 1);
    conv(t128a, nullptr, 0, 4, 128, 128, t128b, 1);
    conv(t128b, nullptr, 0, 5, 128, 128, c2, 1);
    pool(c2, p2, 128, 128, 128);
    // conv3 @64
    conv(p2, nullptr, 0, 6, 64, 64, t64a, 1);
    conv(t64a, nullptr, 0, 7, 64, 64, t64b, 1);
    conv(t64b, nullptr, 0, 8, 64, 64, c3, 1);
    pool(c3, p3, 256, 64, 64);
    // conv4 @32
    conv(p3, nullptr, 0, 9, 32, 32, t32a, 1);
    conv(t32a, nullptr, 0, 10, 32, 32, t32b, 1);
    conv(t32b, nullptr, 0, 11, 32, 32, c4, 1);
    pool(c4, p4, 512, 32, 32);
    // conv5 @16
    conv(p4, nullptr, 0, 12, 16, 16, t16a, 1);
    conv(t16a, nullptr, 0, 13, 16, 16, t16b, 1);
    conv(t16b, nullptr, 0, 14, 16, 16, t16a, 1);    // c5 = t16a
    // decoder: up(c5) + fused concat with c4  (conv6 @32)
    up(t16a, t32a, 512, 16, 16);                    // u5 = t32a
    conv(c4, t32a, 512, 15, 32, 32, t32b, 1);
    conv(t32b, nullptr, 0, 16, 32, 32, t32a, 1);
    conv(t32a, nullptr, 0, 17, 32, 32, t32b, 1);    // c6 = t32b
    // conv7 @64 (concat c3 || up(c6))
    up(t32b, u6, 512, 32, 32);
    conv(c3, u6, 256, 18, 64, 64, t64a, 1);
    conv(t64a, nullptr, 0, 19, 64, 64, t64b, 1);
    conv(t64b, nullptr, 0, 20, 64, 64, t64a, 1);    // c7 = t64a
    // conv8 @128 (concat c2 || up(c7))
    up(t64a, u7, 256, 64, 64);
    conv(c2, u7, 128, 21, 128, 128, t128a, 1);
    conv(t128a, nullptr, 0, 22, 128, 128, t128b, 1);
    conv(t128b, nullptr, 0, 23, 128, 128, t128a, 1); // c8 = t128a
    // outc 1x1 @256 on up(c8), no ReLU -> core
    up(t128a, u8, 75, 128, 128);
    conv(u8, nullptr, 0, 24, 256, 256, core, 0);
    // final per-pixel dynamic 5x5 conv -> d_out
    {
        size_t n = (size_t)B * 3 * 256 * 256;
        kpn_apply_kernel<<<(unsigned)((n + 255) / 256), 256, 0, stream>>>(
            data, core, (float*)d_out, B, 3, 256, 256);
    }
}
